// LSTMModel_47012712022524
// MI455X (gfx1250) — matrix operand order repaired and checked by reading, not compiled, not run on hardware
//
#include <hip/hip_runtime.h>

typedef __attribute__((ext_vector_type(16))) _Float16 v16h;
typedef __attribute__((ext_vector_type(8)))  float    v8f;

#define H     128
#define NL    3
#define VOCAB 256
#define GATES 512   // 4*H

__device__ __forceinline__ float sigmoidf_(float x) {
  return 1.0f / (1.0f + __expf(-x));
}

// Load 8 consecutive f32 (32B aligned) and convert into a[off..off+7] as f16.
__device__ __forceinline__ void load8cvt(const float* __restrict__ p, v16h& a, const int off) {
  const float4 lo = *reinterpret_cast<const float4*>(p);
  const float4 hi = *reinterpret_cast<const float4*>(p + 4);
  a[off + 0] = (_Float16)lo.x; a[off + 1] = (_Float16)lo.y;
  a[off + 2] = (_Float16)lo.z; a[off + 3] = (_Float16)lo.w;
  a[off + 4] = (_Float16)hi.x; a[off + 5] = (_Float16)hi.y;
  a[off + 6] = (_Float16)hi.z; a[off + 7] = (_Float16)hi.w;
}

__global__ __launch_bounds__(512, 1)
void lstm_step_kernel(const int*   __restrict__ x,
                      const float* __restrict__ h0,
                      const float* __restrict__ c0,
                      const float* __restrict__ emb,
                      const float* __restrict__ w_ih,
                      const float* __restrict__ w_hh,
                      const float* __restrict__ b_ih,
                      const float* __restrict__ b_hh,
                      const float* __restrict__ Wd,
                      const float* __restrict__ bd,
                      float*       __restrict__ out)
{
  __shared__ float uvec[2 * H];   // [ x_in (or h from layer below) | h0[l] ]
  __shared__ float gbuf[GATES];   // pre-activation gates

  const int tid  = threadIdx.x;   // 0..511  (16 waves of 32)
  const int lane = tid & 31;
  const int wv   = tid >> 5;      // wave id 0..15
  const int mrow = lane & 15;     // row within 16-row tile
  const int half = lane >> 4;     // 0 or 1 (selects k-group / output row half)

  // ---- Phase A: encoder (embedding lookup) + layer-0 hidden state into LDS
  const int tok = x[0];
  if (tid < H)              uvec[tid] = emb[tok * H + tid];
  else if (tid < 2 * H)     uvec[tid] = h0[tid - H];
  __syncthreads();

  for (int l = 0; l < NL; ++l) {
    const float* wih = w_ih + (size_t)l * GATES * H;
    const float* whh = w_hh + (size_t)l * GATES * H;

    // Prefetch next layer's weights (or decoder) to hide HBM latency.
    if (l + 1 < NL) {
      const float* nih = wih + GATES * H;
      const float* nhh = whh + GATES * H;
      #pragma unroll
      for (int s = 0; s < 2; ++s) {
        __builtin_prefetch(nih + (tid + s * 512) * 64, 0, 3);
        __builtin_prefetch(nhh + (tid + s * 512) * 64, 0, 3);
      }
    } else {
      __builtin_prefetch(Wd + tid * 64, 0, 3);
    }

    // ---- Phase B: gates[512] = [W_ih | W_hh] @ [x_in ; h_prev]  via WMMA
    // 32 row-tiles of 16 rows, k = 256 in 8 chunks of 32. B columns replicated.
    for (int t = wv; t < GATES / 16; t += 16) {
      v8f acc = {};
      const int m = t * 16 + mrow;
      #pragma unroll
      for (int kc = 0; kc < 8; ++kc) {
        const int k0 = kc * 32 + half * 8;   // first 8-run of k for this lane
        const int k1 = k0 + 16;              // second 8-run
        v16h a;
        {
          const float* p0 = (k0 < H) ? (wih + m * H + k0) : (whh + m * H + (k0 - H));
          load8cvt(p0, a, 0);
        }
        {
          const float* p1 = (k1 < H) ? (wih + m * H + k1) : (whh + m * H + (k1 - H));
          load8cvt(p1, a, 8);
        }
        v16h b;
        const int kb = kc * 32 + half * 8;   // two 8-runs of k for B (kb, kb+16)
        #pragma unroll
        for (int i = 0; i < 16; ++i) b[i] = (_Float16)uvec[kb + (i < 8 ? i : i + 8)];
        acc = __builtin_amdgcn_wmma_f32_16x16x32_f16(false, a, false, b,
                                                     (short)0, acc, false, false);
      }
      // D layout: lanes with mrow==0 hold rows (half*8 + j) of this tile.
      if (mrow == 0) {
        #pragma unroll
        for (int j = 0; j < 8; ++j) gbuf[t * 16 + half * 8 + j] = acc[j];
      }
    }
    __syncthreads();

    // ---- Phase C: biases + activations + state update (PyTorch gate order i,f,g,o)
    if (tid < H) {
      const int gb = l * 4 * H;
      const float gi = gbuf[tid]           + b_ih[gb + tid]           + b_hh[gb + tid];
      const float gf = gbuf[H + tid]       + b_ih[gb + H + tid]       + b_hh[gb + H + tid];
      const float gg = gbuf[2 * H + tid]   + b_ih[gb + 2 * H + tid]   + b_hh[gb + 2 * H + tid];
      const float go = gbuf[3 * H + tid]   + b_ih[gb + 3 * H + tid]   + b_hh[gb + 3 * H + tid];
      const float cc = sigmoidf_(gf) * c0[l * H + tid] + sigmoidf_(gi) * tanhf(gg);
      const float hh = sigmoidf_(go) * tanhf(cc);
      out[VOCAB + NL * H + l * H + tid] = cc;   // cn  (offset 640)
      out[VOCAB + l * H + tid]          = hh;   // hn  (offset 256)
      uvec[tid] = hh;                           // input to next layer / decoder
    } else if (tid < 2 * H && (l + 1) < NL) {
      uvec[tid] = h0[(l + 1) * H + (tid - H)];  // next layer's hidden state
    }
    __syncthreads();
  }

  // ---- Decoder: logits[256] = Wd @ h2 + bd.  16 tiles, one per wave, k = 128.
  {
    const int t = wv;
    v8f acc = {};
    const int m = t * 16 + mrow;
    #pragma unroll
    for (int kc = 0; kc < 4; ++kc) {
      const int k0 = kc * 32 + half * 8;
      v16h a;
      load8cvt(Wd + m * H + k0,      a, 0);
      load8cvt(Wd + m * H + k0 + 16, a, 8);
      v16h b;
      const int kb = kc * 32 + half * 8;
      #pragma unroll
      for (int i = 0; i < 16; ++i) b[i] = (_Float16)uvec[kb + (i < 8 ? i : i + 8)];
      acc = __builtin_amdgcn_wmma_f32_16x16x32_f16(false, a, false, b,
                                                   (short)0, acc, false, false);
    }
    if (mrow == 0) {
      #pragma unroll
      for (int j = 0; j < 8; ++j) {
        const int row = t * 16 + half * 8 + j;
        out[row] = acc[j] + bd[row];
      }
    }
  }
}

extern "C" void kernel_launch(void* const* d_in, const int* in_sizes, int n_in,
                              void* d_out, int out_size, void* d_ws, size_t ws_size,
                              hipStream_t stream) {
  const int*   x    = (const int*)  d_in[0];
  const float* h0   = (const float*)d_in[1];
  const float* c0   = (const float*)d_in[2];
  const float* emb  = (const float*)d_in[3];
  const float* w_ih = (const float*)d_in[4];
  const float* w_hh = (const float*)d_in[5];
  const float* b_ih = (const float*)d_in[6];
  const float* b_hh = (const float*)d_in[7];
  const float* Wd   = (const float*)d_in[8];
  const float* bd   = (const float*)d_in[9];
  float* out = (float*)d_out;

  lstm_step_kernel<<<1, 512, 0, stream>>>(x, h0, c0, emb, w_ih, w_hh,
                                          b_ih, b_hh, Wd, bd, out);
}
